// SAGE_2405181685928
// MI455X (gfx1250) — compile-verified, matrix operand order checked by reading, not run on hardware
//
#include <hip/hip_runtime.h>

typedef __attribute__((ext_vector_type(16))) _Float16 v16h;
typedef __attribute__((ext_vector_type(8)))  float    v8f;

#define NN      16384
#define INC     128
#define HIDC    128
#define OUTC    64
#define NE      524288
#define NLBL    131072

// ---------------- WMMA helpers ----------------

__device__ __forceinline__ v8f wmma_f16(v16h a, v16h b, v8f c) {
  return __builtin_amdgcn_wmma_f32_16x16x32_f16(
      false, a, false, b, (short)0, c, false, false);
}

// 16-bit A/B fragment (16x32 MxK), ISA layout:
// lane = h*16 + r ; VGPR i holds K = k0 + 16*(i>>2) + 2*(i&3) + 8*h (pairs contiguous)
__device__ __forceinline__ v16h frag_f16(const _Float16* p, int ld, int row0,
                                         int k0, int lane) {
  int r = lane & 15, h = lane >> 4;
  const _Float16* base = p + (size_t)(row0 + r) * ld + k0 + 8 * h;
  v16h f;
#pragma unroll
  for (int i = 0; i < 8; ++i) {
    int k = 16 * (i >> 2) + 2 * (i & 3);
    f[2 * i]     = base[k];
    f[2 * i + 1] = base[k + 1];
  }
  return f;
}

__device__ __forceinline__ v16h frag_f32(const float* p, int ld, int row0,
                                         int k0, int lane) {
  int r = lane & 15, h = lane >> 4;
  const float* base = p + (size_t)(row0 + r) * ld + k0 + 8 * h;
  v16h f;
#pragma unroll
  for (int i = 0; i < 8; ++i) {
    int k = 16 * (i >> 2) + 2 * (i & 3);
    f[2 * i]     = (_Float16)base[k];
    f[2 * i + 1] = (_Float16)base[k + 1];
  }
  return f;
}

// ---------------- small utility kernels ----------------

__global__ void zero_f32(float* p, int n) {
  int i = blockIdx.x * blockDim.x + threadIdx.x;
  if (i < n) p[i] = 0.0f;
}

// W[k][n] f32 -> WT[n][k] f16
__global__ void transpose_w(const float* __restrict__ W, _Float16* __restrict__ WT,
                            int K, int N) {
  int t = blockIdx.x * blockDim.x + threadIdx.x;
  if (t >= K * N) return;
  int n = t / K, k = t % K;
  WT[n * K + k] = (_Float16)W[k * N + n];
}

__global__ void deg_kernel(const int* __restrict__ dst, float* deg, int E) {
  int e = blockIdx.x * blockDim.x + threadIdx.x;
  if (e < E) atomicAdd(&deg[dst[e]], 1.0f);
}

// 32 lanes per edge, each lane scatters 4 channels (C == 128)
__global__ void scatter_kernel(const float* __restrict__ feat,
                               const int* __restrict__ src,
                               const int* __restrict__ dst,
                               float* __restrict__ agg, int E) {
  int t = blockIdx.x * blockDim.x + threadIdx.x;
  int e = t >> 5;
  if (e >= E) return;
  int c4 = (t & 31) * 4;
  int s = src[e], d = dst[e];
  float4 v = *(const float4*)(feat + (size_t)s * 128 + c4);
  float* o = agg + (size_t)d * 128 + c4;
  atomicAdd(o + 0, v.x);
  atomicAdd(o + 1, v.y);
  atomicAdd(o + 2, v.z);
  atomicAdd(o + 3, v.w);
}

__global__ void divide_kernel(float* __restrict__ agg,
                              const float* __restrict__ deg, int total) {
  int i = blockIdx.x * blockDim.x + threadIdx.x;
  if (i >= total) return;
  agg[i] /= fmaxf(deg[i >> 7], 1.0f);
}

// ---------------- layer GEMMs ----------------

// h1 = relu(agg @ Wl + b + x @ Wr)   [NN, 128], K = 128
__global__ void gemm1_kernel(const float* __restrict__ agg,
                             const float* __restrict__ x,
                             const _Float16* __restrict__ WTl,
                             const _Float16* __restrict__ WTr,
                             const float* __restrict__ bias,
                             float* __restrict__ h1) {
  int lane = threadIdx.x;
  int tile = blockIdx.x * blockDim.y + threadIdx.y;  // 8192 tiles
  int tm = tile >> 3, tn = tile & 7;
  v8f acc = {};
#pragma unroll
  for (int kb = 0; kb < 128; kb += 32) {
    v16h a1 = frag_f32(agg, 128, tm * 16, kb, lane);
    v16h b1 = frag_f16(WTl, 128, tn * 16, kb, lane);
    acc = wmma_f16(a1, b1, acc);
    v16h a2 = frag_f32(x, 128, tm * 16, kb, lane);
    v16h b2 = frag_f16(WTr, 128, tn * 16, kb, lane);
    acc = wmma_f16(a2, b2, acc);
  }
  int n = tn * 16 + (lane & 15);
  int h = lane >> 4;
  float bv = bias[n];
#pragma unroll
  for (int r = 0; r < 8; ++r) {
    int m = tm * 16 + r + 8 * h;
    float v = acc[r] + bv;
    h1[(size_t)m * 128 + n] = v > 0.0f ? v : 0.0f;
  }
}

// z = agg2 @ Wl2 + b2 + h1 @ Wr2    [NN, 64], K = 128 ; also emits zh (f16)
__global__ void gemm2_kernel(const float* __restrict__ agg,
                             const float* __restrict__ h1,
                             const _Float16* __restrict__ WTl,
                             const _Float16* __restrict__ WTr,
                             const float* __restrict__ bias,
                             float* __restrict__ z,
                             _Float16* __restrict__ zh) {
  int lane = threadIdx.x;
  int tile = blockIdx.x * blockDim.y + threadIdx.y;  // 4096 tiles
  int tm = tile >> 2, tn = tile & 3;
  v8f acc = {};
#pragma unroll
  for (int kb = 0; kb < 128; kb += 32) {
    v16h a1 = frag_f32(agg, 128, tm * 16, kb, lane);
    v16h b1 = frag_f16(WTl, 128, tn * 16, kb, lane);
    acc = wmma_f16(a1, b1, acc);
    v16h a2 = frag_f32(h1, 128, tm * 16, kb, lane);
    v16h b2 = frag_f16(WTr, 128, tn * 16, kb, lane);
    acc = wmma_f16(a2, b2, acc);
  }
  int n = tn * 16 + (lane & 15);
  int h = lane >> 4;
  float bv = bias[n];
#pragma unroll
  for (int r = 0; r < 8; ++r) {
    int m = tm * 16 + r + 8 * h;
    float v = acc[r] + bv;
    z[(size_t)m * 64 + n]  = v;
    zh[(size_t)m * 64 + n] = (_Float16)v;
  }
}

// ---------------- decode (wave per label edge) ----------------

__global__ void decode_kernel(const float* __restrict__ z,
                              const int* __restrict__ eli,
                              float* __restrict__ dmul,
                              float* __restrict__ dsum, int E) {
  int t = blockIdx.x * blockDim.x + threadIdx.x;
  int e = t >> 5;
  if (e >= E) return;
  int lane = t & 31;
  int a = eli[e], b = eli[E + e];
  float2 za = *(const float2*)(z + (size_t)a * 64 + lane * 2);
  float2 zb = *(const float2*)(z + (size_t)b * 64 + lane * 2);
  float m0 = za.x * zb.x, m1 = za.y * zb.y;
  float2 m = {m0, m1};
  *(float2*)(dmul + (size_t)e * 64 + lane * 2) = m;
  float s = m0 + m1;
#pragma unroll
  for (int off = 16; off > 0; off >>= 1) s += __shfl_xor(s, off, 32);
  if (lane == 0) dsum[e] = s;
}

// ---------------- z @ z^T + mask (store-bandwidth bound) ----------------
// B = z^T, so the B fragment for output-column block tj is just rows of z:
// both operands are contiguous row loads from the 2 MB f16 copy (L2 resident).
__global__ void zzT_kernel(const _Float16* __restrict__ zh,
                           float* __restrict__ padj,
                           float* __restrict__ mask) {
  int lane = threadIdx.x;
  int tj = blockIdx.x * blockDim.y + threadIdx.y;  // 0..1023 (N tiles)
  int ti = blockIdx.y;                             // 0..1023 (M tiles)
  v8f acc = {};
  v16h a0 = frag_f16(zh, 64, ti * 16, 0, lane);
  v16h b0 = frag_f16(zh, 64, tj * 16, 0, lane);
  acc = wmma_f16(a0, b0, acc);
  v16h a1 = frag_f16(zh, 64, ti * 16, 32, lane);
  v16h b1 = frag_f16(zh, 64, tj * 16, 32, lane);
  acc = wmma_f16(a1, b1, acc);
  int n = tj * 16 + (lane & 15);
  int h = lane >> 4;
#pragma unroll
  for (int r = 0; r < 8; ++r) {
    size_t m = (size_t)(ti * 16 + r + 8 * h);
    float v = acc[r];
    __builtin_nontemporal_store(v, padj + m * NN + n);
    __builtin_nontemporal_store(v > 0.0f ? 1.0f : 0.0f, mask + m * NN + n);
  }
}

// ---------------- launch ----------------

extern "C" void kernel_launch(void* const* d_in, const int* in_sizes, int n_in,
                              void* d_out, int out_size, void* d_ws, size_t ws_size,
                              hipStream_t stream) {
  const float* x    = (const float*)d_in[0];
  const int*   ei   = (const int*)d_in[1];   // [2, NE]
  const int*   eli  = (const int*)d_in[2];   // [2, NLBL]
  const float* W_l1 = (const float*)d_in[3];
  const float* b_l1 = (const float*)d_in[4];
  const float* W_r1 = (const float*)d_in[5];
  const float* W_l2 = (const float*)d_in[6];
  const float* b_l2 = (const float*)d_in[7];
  const float* W_r2 = (const float*)d_in[8];
  const int* src = ei;
  const int* dst = ei + NE;

  // output layout (flat f32, return-tuple order)
  float* out = (float*)d_out;
  float* o_sage1 = out;                                  // NN*128
  float* o_z     = o_sage1 + (size_t)NN * HIDC;          // NN*64
  float* o_dmul  = o_z + (size_t)NN * OUTC;              // NLBL*64
  float* o_dsum  = o_dmul + (size_t)NLBL * OUTC;         // NLBL
  float* o_padj  = o_dsum + (size_t)NLBL;                // NN*NN
  float* o_mask  = o_padj + (size_t)NN * NN;             // NN*NN

  // workspace layout
  float* agg1 = (float*)d_ws;                            // NN*128
  float* agg2 = agg1 + (size_t)NN * HIDC;                // NN*128
  float* deg  = agg2 + (size_t)NN * HIDC;                // NN
  _Float16* zh   = (_Float16*)(deg + NN);                // NN*64
  _Float16* WTl1 = zh + (size_t)NN * OUTC;               // 128*128
  _Float16* WTr1 = WTl1 + INC * HIDC;
  _Float16* WTl2 = WTr1 + INC * HIDC;                    // 64*128
  _Float16* WTr2 = WTl2 + HIDC * OUTC;

  const int ZN = 2 * NN * HIDC + NN;
  zero_f32<<<(ZN + 255) / 256, 256, 0, stream>>>(agg1, ZN);

  transpose_w<<<(INC * HIDC + 255) / 256, 256, 0, stream>>>(W_l1, WTl1, INC, HIDC);
  transpose_w<<<(INC * HIDC + 255) / 256, 256, 0, stream>>>(W_r1, WTr1, INC, HIDC);
  transpose_w<<<(HIDC * OUTC + 255) / 256, 256, 0, stream>>>(W_l2, WTl2, HIDC, OUTC);
  transpose_w<<<(HIDC * OUTC + 255) / 256, 256, 0, stream>>>(W_r2, WTr2, HIDC, OUTC);

  deg_kernel<<<NE / 256, 256, 0, stream>>>(dst, deg, NE);
  scatter_kernel<<<(NE * 32) / 256, 256, 0, stream>>>(x, src, dst, agg1, NE);
  divide_kernel<<<(NN * HIDC) / 256, 256, 0, stream>>>(agg1, deg, NN * HIDC);

  // layer 1: 8192 wave-tiles, 8 waves/block
  gemm1_kernel<<<1024, dim3(32, 8), 0, stream>>>(agg1, x, WTl1, WTr1, b_l1, o_sage1);

  scatter_kernel<<<(NE * 32) / 256, 256, 0, stream>>>(o_sage1, src, dst, agg2, NE);
  divide_kernel<<<(NN * HIDC) / 256, 256, 0, stream>>>(agg2, deg, NN * HIDC);

  // layer 2: 4096 wave-tiles
  gemm2_kernel<<<512, dim3(32, 8), 0, stream>>>(agg2, o_sage1, WTl2, WTr2, b_l2,
                                                o_z, zh);

  decode_kernel<<<(NLBL * 32) / 256, 256, 0, stream>>>(o_z, eli, o_dmul, o_dsum, NLBL);

  // 1024x1024 output tiles, 8 waves/block along N
  zzT_kernel<<<dim3(128, 1024), dim3(32, 8), 0, stream>>>(zh, o_padj, o_mask);
}
